// Net_41575283425958
// MI455X (gfx1250) — matrix operand order repaired and checked by reading, compiled, not run on hardware
//
#include <hip/hip_runtime.h>
#include <hip/hip_bf16.h>
#include <math.h>

// ---------------------------------------------------------------------------
// GNN pipeline for MI455X (gfx1250, wave32, WMMA).
//   h0 = sum of 9 embedding gathers               [N,128] f16
//   agg1 = segment_max_{src->dst} h0              (sortable-u32 atomic max)
//   h1 = relu(agg1@W1_rel + h0@W1_root + b1)      [N,64]  f16  (WMMA f16->f32)
//   agg2 = segment_max h1
//   h2 = relu(agg2@W2_rel + h1@W2_root + b2)      [N,32]  f16  (WMMA)
//   g  = segment_sum_{batch} h2                   [G,32]  f32  (atomic add)
//   out = relu(g@Wl1+bl1)@Wl2+bl2                 [G,2]   f32
// ---------------------------------------------------------------------------

typedef __attribute__((ext_vector_type(16))) _Float16 v16h;
typedef __attribute__((ext_vector_type(8)))  _Float16 h8v;     // 16B
typedef __attribute__((ext_vector_type(8)))  float    v8f;
typedef __attribute__((ext_vector_type(4)))  unsigned u4v;     // 16B

#define HID 128

// ---- order-preserving float <-> u32 (native global_atomic_max_u32 path) ----
__device__ __forceinline__ unsigned enc_sortable(float x) {
    unsigned b = __float_as_uint(x);
    return b ^ ((b >> 31) ? 0xFFFFFFFFu : 0x80000000u);   // any finite -> > 0
}
__device__ __forceinline__ float dec_sortable(unsigned u) {
    if (u == 0u) return 0.0f;                              // empty segment -> 0 (PyG fill)
    unsigned b = (u & 0x80000000u) ? (u ^ 0x80000000u) : ~u;
    return __uint_as_float(b);
}

// ---------------------------------------------------------------------------
__global__ void fill_zero_u32(unsigned* __restrict__ p, size_t n) {
    size_t i = (size_t)blockIdx.x * blockDim.x + threadIdx.x;
    if (i < n) p[i] = 0u;
}

// ---- AtomEncoder: h0[n,c] = sum_t table_t[x[n,t]][c], stored f16 -----------
__global__ void encode_kernel(const int* __restrict__ x,
                              const float* __restrict__ t0, const float* __restrict__ t1,
                              const float* __restrict__ t2, const float* __restrict__ t3,
                              const float* __restrict__ t4, const float* __restrict__ t5,
                              const float* __restrict__ t6, const float* __restrict__ t7,
                              const float* __restrict__ t8,
                              _Float16* __restrict__ h0, int N) {
    size_t i = (size_t)blockIdx.x * blockDim.x + threadIdx.x;
    if (i >= (size_t)N * HID) return;
    int n = (int)(i >> 7);        // /128
    int c = (int)(i & 127);
    const int* xr = x + (size_t)n * 9;   // same row for 128 consecutive lanes -> cache hit
    float s = t0[(size_t)xr[0] * HID + c];
    s += t1[(size_t)xr[1] * HID + c];
    s += t2[(size_t)xr[2] * HID + c];
    s += t3[(size_t)xr[3] * HID + c];
    s += t4[(size_t)xr[4] * HID + c];
    s += t5[(size_t)xr[5] * HID + c];
    s += t6[(size_t)xr[6] * HID + c];
    s += t7[(size_t)xr[7] * HID + c];
    s += t8[(size_t)xr[8] * HID + c];
    h0[i] = (_Float16)s;
}

// ---- edge scatter-max: one wave per edge, lanes sweep the feature dim ------
template <int H>
__global__ void scatter_max_kernel(const _Float16* __restrict__ h,
                                   const int* __restrict__ src,
                                   const int* __restrict__ dst,
                                   unsigned* __restrict__ agg, int E) {
    size_t gtid = (size_t)blockIdx.x * blockDim.x + threadIdx.x;
    int e = (int)(gtid >> 5);
    if (e >= E) return;
    int lane = (int)(gtid & 31);
    int s = src[e], d = dst[e];
    const _Float16* hp = h + (size_t)s * H;
    unsigned* ap = agg + (size_t)d * H;
#pragma unroll
    for (int j = 0; j < H / 32; ++j) {
        int c = lane + 32 * j;                 // coalesced 2B loads / 4B atomics
        float v = (float)hp[c];
        atomicMax(&ap[c], enc_sortable(v));    // global_atomic_max_u32
    }
}

// ---- fused GraphConv GEMM: relu(dec(agg)@W_rel + h@W_root + b) -> f16 ------
// Per wave: one 16-node tile, NC/16 column tiles, K/32 k-steps, WMMA f16->f32.
// Weights staged TRANSPOSED ([NC][K] f16) in LDS so each lane's B fragment is two
// contiguous 8-half k-runs -> two ds_load_b128 per fragment.
template <int K, int NC>
__global__ void graphconv_wmma_kernel(const unsigned* __restrict__ agg_u,
                                      const _Float16* __restrict__ hroot,
                                      const float* __restrict__ Wrel,
                                      const float* __restrict__ Wroot,
                                      const float* __restrict__ bias,
                                      _Float16* __restrict__ out, int N) {
    __shared__ _Float16 sWt[2 * K * NC];       // [rel^T | root^T], each [NC][K]

    // cooperative f32->f16 transpose staging (global reads coalesced)
    for (int i = threadIdx.x; i < K * NC; i += blockDim.x) {
        int k = i / NC;            // NC is a power of two -> shifts
        int c = i - k * NC;
        sWt[c * K + k]          = (_Float16)Wrel[i];
        sWt[K * NC + c * K + k] = (_Float16)Wroot[i];
    }
    __syncthreads();

    const int lane = threadIdx.x & 31;
    const int wave = threadIdx.x >> 5;
    const int ntiles = (N + 15) >> 4;
    const int tile = blockIdx.x * (blockDim.x >> 5) + wave;
    if (tile >= ntiles) return;

    const int row_base = tile * 16;
    const int col  = lane & 15;       // B/C/D: column within 16-wide tile
    const int half = lane >> 4;       // K-half selector
    const int arow = lane & 15;       // A: matrix row
    const int ab   = 8 * half;        // A: K sub-base (ISA 16-bit A layout)
    const int kb   = 8 * half;        // B: K sub-base (same table order as A)

    v8f acc[NC / 16];
#pragma unroll
    for (int ct = 0; ct < NC / 16; ++ct) {
        float bv = bias[ct * 16 + col];
#pragma unroll
        for (int r = 0; r < 8; ++r) acc[ct][r] = bv;
    }

    int row = row_base + arow;
    if (row >= N) row = N - 1;        // clamp loads; stores guarded below

#pragma unroll
    for (int ks = 0; ks < K / 32; ++ks) {
        // A (rel): vector-load sortable-u32 segment-max, decode+cvt to f16
        v16h a_rel;
        {
            const u4v* up = (const u4v*)(agg_u + (size_t)row * K + ks * 32 + ab);
            u4v l0 = up[0], l1 = up[1];        // K = ab .. ab+7
            u4v h0_ = up[4], h1 = up[5];       // K = ab+16 .. ab+23
#pragma unroll
            for (int i = 0; i < 4; ++i) {
                a_rel[i]      = (_Float16)dec_sortable(l0[i]);
                a_rel[4 + i]  = (_Float16)dec_sortable(l1[i]);
                a_rel[8 + i]  = (_Float16)dec_sortable(h0_[i]);
                a_rel[12 + i] = (_Float16)dec_sortable(h1[i]);
            }
        }
        // A (root): straight f16 loads, two 16B beats
        v16h a_root;
        {
            const _Float16* rp = hroot + (size_t)row * K + ks * 32 + ab;
            h8v lo = *(const h8v*)rp;
            h8v hi = *(const h8v*)(rp + 16);
#pragma unroll
            for (int i = 0; i < 8; ++i) { a_root[i] = lo[i]; a_root[8 + i] = hi[i]; }
        }
#pragma unroll
        for (int ct = 0; ct < NC / 16; ++ct) {
            // B fragments: two 8-half runs (k = kb.., kb+16..) per lane from transposed LDS
            const _Float16* brel = sWt + (ct * 16 + col) * K + ks * 32 + kb;
            const _Float16* brt  = brel + K * NC;
            h8v r0 = *(const h8v*)brel, r1 = *(const h8v*)(brel + 16);
            h8v t0 = *(const h8v*)brt,  t1 = *(const h8v*)(brt + 16);
            v16h b_rel, b_root;
#pragma unroll
            for (int i = 0; i < 8; ++i) {
                b_rel[i] = r0[i];  b_rel[8 + i] = r1[i];
                b_root[i] = t0[i]; b_root[8 + i] = t1[i];
            }
            acc[ct] = __builtin_amdgcn_wmma_f32_16x16x32_f16(
                false, a_rel, false, b_rel, (short)0, acc[ct], false, false);
            acc[ct] = __builtin_amdgcn_wmma_f32_16x16x32_f16(
                false, a_root, false, b_root, (short)0, acc[ct], false, false);
        }
    }

    // D layout: lane col = lane&15; VGPR r -> row M = r + 8*(lane>>4)
#pragma unroll
    for (int ct = 0; ct < NC / 16; ++ct) {
#pragma unroll
        for (int r = 0; r < 8; ++r) {
            int node = row_base + r + 8 * half;
            if (node < N) {
                float v = acc[ct][r];
                out[(size_t)node * NC + ct * 16 + col] = (_Float16)fmaxf(v, 0.0f);
            }
        }
    }
}

// ---- global_add_pool: g[batch[n]] += h2[n] ---------------------------------
__global__ void pool_kernel(const _Float16* __restrict__ h2,
                            const int* __restrict__ batch,
                            float* __restrict__ g, int N) {
    size_t i = (size_t)blockIdx.x * blockDim.x + threadIdx.x;
    if (i >= (size_t)N * 32) return;
    int n = (int)(i >> 5);
    int c = (int)(i & 31);
    atomicAdd(&g[(size_t)batch[n] * 32 + c], (float)h2[i]);   // global_atomic_add_f32
}

// ---- head MLP: relu(g@Wl1+bl1)@Wl2+bl2 ------------------------------------
__global__ void mlp_kernel(const float* __restrict__ g,
                           const float* __restrict__ Wl1, const float* __restrict__ bl1,
                           const float* __restrict__ Wl2, const float* __restrict__ bl2,
                           float* __restrict__ out, int G) {
    int i = blockIdx.x * blockDim.x + threadIdx.x;
    if (i >= G) return;
    const float* gr = g + (size_t)i * 32;
    float hid[16];
#pragma unroll
    for (int o = 0; o < 16; ++o) {
        float s = bl1[o];
#pragma unroll
        for (int k = 0; k < 32; ++k) s = fmaf(gr[k], Wl1[k * 16 + o], s);
        hid[o] = fmaxf(s, 0.0f);
    }
#pragma unroll
    for (int o = 0; o < 2; ++o) {
        float s = bl2[o];
#pragma unroll
        for (int k = 0; k < 16; ++k) s = fmaf(hid[k], Wl2[k * 2 + o], s);
        out[(size_t)i * 2 + o] = s;
    }
}

// ---------------------------------------------------------------------------
extern "C" void kernel_launch(void* const* d_in, const int* in_sizes, int n_in,
                              void* d_out, int out_size, void* d_ws, size_t ws_size,
                              hipStream_t stream) {
    (void)n_in; (void)ws_size;
    // dict order: x, edge_index, batch, emb_tables[0..8], W1_rel, W1_root, b1,
    //             W2_rel, W2_root, b2, Wl1, bl1, Wl2, bl2
    const int*   x     = (const int*)d_in[0];
    const int*   ei    = (const int*)d_in[1];
    const int*   batch = (const int*)d_in[2];
    const float* T[9];
    for (int t = 0; t < 9; ++t) T[t] = (const float*)d_in[3 + t];
    const float* W1_rel  = (const float*)d_in[12];
    const float* W1_root = (const float*)d_in[13];
    const float* b1      = (const float*)d_in[14];
    const float* W2_rel  = (const float*)d_in[15];
    const float* W2_root = (const float*)d_in[16];
    const float* b2      = (const float*)d_in[17];
    const float* Wl1     = (const float*)d_in[18];
    const float* bl1     = (const float*)d_in[19];
    const float* Wl2     = (const float*)d_in[20];
    const float* bl2     = (const float*)d_in[21];
    float* out = (float*)d_out;

    const int N = in_sizes[0] / 9;
    const int E = in_sizes[1] / 2;
    const int G = out_size / 2;
    const int* src = ei;
    const int* dst = ei + E;

    // workspace carve-up (256B aligned)
    auto align256 = [](size_t v) { return (v + 255) & ~(size_t)255; };
    char* ws = (char*)d_ws;
    size_t off = 0;
    _Float16* h0   = (_Float16*)(ws + off); off = align256(off + (size_t)N * 128 * sizeof(_Float16));
    unsigned* agg1 = (unsigned*)(ws + off); off = align256(off + (size_t)N * 128 * sizeof(unsigned));
    _Float16* h1   = (_Float16*)(ws + off); off = align256(off + (size_t)N * 64 * sizeof(_Float16));
    unsigned* agg2 = (unsigned*)(ws + off); off = align256(off + (size_t)N * 64 * sizeof(unsigned));
    _Float16* h2   = (_Float16*)(ws + off); off = align256(off + (size_t)N * 32 * sizeof(_Float16));
    float*    g    = (float*)(ws + off);    off = align256(off + (size_t)G * 32 * sizeof(float));

    const int TB = 256;
    // zero atomics targets (agg sentinel 0u == empty; g accumulates f32 zeros)
    {
        size_t n1 = (size_t)N * 128;
        fill_zero_u32<<<(unsigned)((n1 + TB - 1) / TB), TB, 0, stream>>>(agg1, n1);
        size_t n2 = (size_t)N * 64;
        fill_zero_u32<<<(unsigned)((n2 + TB - 1) / TB), TB, 0, stream>>>(agg2, n2);
        size_t n3 = (size_t)G * 32;
        fill_zero_u32<<<(unsigned)((n3 + TB - 1) / TB), TB, 0, stream>>>((unsigned*)g, n3);
    }

    // 1. encoder
    {
        size_t nt = (size_t)N * 128;
        encode_kernel<<<(unsigned)((nt + TB - 1) / TB), TB, 0, stream>>>(
            x, T[0], T[1], T[2], T[3], T[4], T[5], T[6], T[7], T[8], h0, N);
    }
    // 2. layer-1 scatter max (one wave per edge)
    {
        size_t nt = (size_t)E * 32;
        scatter_max_kernel<128><<<(unsigned)((nt + TB - 1) / TB), TB, 0, stream>>>(
            h0, src, dst, agg1, E);
    }
    // 3. GraphConv1 WMMA (128 -> 64), 8 waves (8 node tiles) per block
    {
        int ntiles = (N + 15) / 16;
        int wavesPerBlock = 8;                 // 256 threads
        int blocks = (ntiles + wavesPerBlock - 1) / wavesPerBlock;
        graphconv_wmma_kernel<128, 64><<<blocks, 256, 0, stream>>>(
            agg1, h0, W1_rel, W1_root, b1, h1, N);
    }
    // 4. layer-2 scatter max
    {
        size_t nt = (size_t)E * 32;
        scatter_max_kernel<64><<<(unsigned)((nt + TB - 1) / TB), TB, 0, stream>>>(
            h1, src, dst, agg2, E);
    }
    // 5. GraphConv2 WMMA (64 -> 32)
    {
        int ntiles = (N + 15) / 16;
        int wavesPerBlock = 8;
        int blocks = (ntiles + wavesPerBlock - 1) / wavesPerBlock;
        graphconv_wmma_kernel<64, 32><<<blocks, 256, 0, stream>>>(
            agg2, h1, W2_rel, W2_root, b2, h2, N);
    }
    // 6. pool
    {
        size_t nt = (size_t)N * 32;
        pool_kernel<<<(unsigned)((nt + TB - 1) / TB), TB, 0, stream>>>(h2, batch, g, N);
    }
    // 7. head MLP
    mlp_kernel<<<(G + TB - 1) / TB, TB, 0, stream>>>(g, Wl1, bl1, Wl2, bl2, out, G);
}
